// Attention_5428838662224
// MI455X (gfx1250) — matrix operand order repaired and checked by reading, not compiled, not run on hardware
//
#include <hip/hip_runtime.h>
#include <hip/hip_bf16.h>

// ---------------------------------------------------------------------------
// Attention with softmax over the QUERY axis (dim=1 of [B, Nq, Nk]).
//   S = Q K^T / sqrt(D);  attn[:,q,k] = exp(S[q,k]-m_k)/Z_k  (m_k,Z_k over q)
//   out = attn @ V
// Strategy (no 256MB score materialization):
//   k1: f32 -> f16 downconvert of Q,K,V into workspace
//   k2: per-column (per-k) online-softmax stats m_k, Z_k via WMMA QK^T tiles
//   k3: recompute S tiles, P = exp(S-m)/Z, out += P @ V via WMMA,
//       double-buffered LDS staging (async global->LDS when available)
// ---------------------------------------------------------------------------

#define BATCH 4
#define SEQ   4096
#define DIM   128
#define SCALE 0.08838834764831845f   // 1/sqrt(128)
#define WPB   4                       // waves per 128-thread block
#define TPB   (WPB * 32)

typedef __attribute__((ext_vector_type(16))) _Float16       v16h;
typedef __attribute__((ext_vector_type(8)))  _Float16       v8h;
typedef __attribute__((ext_vector_type(8)))  float          v8f;
typedef __attribute__((ext_vector_type(8)))  unsigned short ushort8;
typedef __attribute__((ext_vector_type(4)))  int            i32x4;

#define AS1 __attribute__((address_space(1)))
#define AS3 __attribute__((address_space(3)))

#if defined(__has_builtin)
#if __has_builtin(__builtin_amdgcn_global_load_async_to_lds_b128)
#define USE_ASYNC_LDS 1
#endif
#endif
#ifndef USE_ASYNC_LDS
#define USE_ASYNC_LDS 0
#endif

__device__ __forceinline__ void wait_async0() {
#if __has_builtin(__builtin_amdgcn_s_wait_asynccnt)
  __builtin_amdgcn_s_wait_asynccnt(0);
#else
  asm volatile("s_wait_asynccnt 0x0" ::: "memory");
#endif
}

// ---- fragment loaders (wave32 WMMA operands, contiguous k order in A and B) -----

// A-matrix 16x32 f16: lane m=lane&15, half h: element e = K(16h + e):
// 16 contiguous f16 (two 16-byte chunks).
template <typename P>
__device__ __forceinline__ v16h load_a_frag(P rowbase, int half) {
  v8h lo = *(const v8h*)(rowbase + 16 * half);
  v8h hi = *(const v8h*)(rowbase + 16 * half + 8);
  v16h r;
#pragma unroll
  for (int i = 0; i < 8; ++i) { r[i] = lo[i]; r[i + 8] = hi[i]; }
  return r;
}

// B-matrix 32x16 f16: lane n=lane&15, half h: element e = K(16h + e):
// 16 contiguous f16 (32 bytes).
template <typename P>
__device__ __forceinline__ v16h load_b_frag(P p) {
  v8h lo = *(const v8h*)(p);
  v8h hi = *(const v8h*)(p + 8);
  v16h r;
#pragma unroll
  for (int i = 0; i < 8; ++i) { r[i] = lo[i]; r[i + 8] = hi[i]; }
  return r;
}

// ---- kernel 1: downconvert ------------------------------------------------
__global__ void attn_cvt_f16(const float* __restrict__ q,
                             const float* __restrict__ k,
                             const float* __restrict__ v,
                             _Float16* __restrict__ qh,
                             _Float16* __restrict__ kh,
                             _Float16* __restrict__ vh, int n) {
  int i = blockIdx.x * blockDim.x + threadIdx.x;
  int stride = gridDim.x * blockDim.x;
  for (; i < n; i += stride) {
    qh[i] = (_Float16)q[i];
    kh[i] = (_Float16)k[i];
    vh[i] = (_Float16)v[i];
  }
}

// ---- kernel 2: per-k-column softmax stats (max & sum over q) --------------
// grid = BATCH * (SEQ/64); 4 waves/block, each wave owns 16 k-columns and
// streams all q tiles. K^T B-fragments are loop-invariant registers.
__global__ __launch_bounds__(TPB) void attn_colstats(
    const _Float16* __restrict__ Qh, const _Float16* __restrict__ Kh,
    float* __restrict__ mOut, float* __restrict__ zOut) {
  const int nKblk = SEQ / (WPB * 16);
  const int b    = blockIdx.x / nKblk;
  const int kblk = blockIdx.x % nKblk;
  const int wave = threadIdx.x >> 5;
  const int lane = threadIdx.x & 31;
  const int half = lane >> 4;
  const int ln   = lane & 15;
  const int k0   = kblk * (WPB * 16) + wave * 16;

  const _Float16* Qb = Qh + (size_t)b * SEQ * DIM;
  const _Float16* Kb = Kh + (size_t)b * SEQ * DIM;

  // B = K^T (32d x 16k) fragments: lane = column k0+ln, contiguous in d.
  v16h bK[4];
#pragma unroll
  for (int c = 0; c < 4; ++c)
    bK[c] = load_b_frag(Kb + (size_t)(k0 + ln) * DIM + 32 * c + 16 * half);

  float mrun = -__builtin_inff();
  float zrun = 0.0f;

  for (int q0 = 0; q0 < SEQ; q0 += 16) {
    if (q0 + 16 < SEQ) {  // prefetch next q tile (global_prefetch_b8)
      const _Float16* np = Qb + (size_t)(q0 + 16 + ln) * DIM;
      __builtin_prefetch(np, 0, 3);
      __builtin_prefetch(np + 64, 0, 3);
    }
    v8f acc = {};
#pragma unroll
    for (int c = 0; c < 4; ++c) {
      v16h aQ = load_a_frag(Qb + (size_t)(q0 + ln) * DIM + 32 * c, half);
      acc = __builtin_amdgcn_wmma_f32_16x16x32_f16(
          false, aQ, false, bK[c], (short)0, acc, false, false);
    }
    // branchless online softmax update for this lane's 8 rows of column ln
    float s[8], cmax = mrun;
#pragma unroll
    for (int r = 0; r < 8; ++r) { s[r] = acc[r] * SCALE; cmax = fmaxf(cmax, s[r]); }
    zrun *= __expf(mrun - cmax);   // mrun == -inf -> factor 0, zrun already 0
    mrun  = cmax;
#pragma unroll
    for (int r = 0; r < 8; ++r) zrun += __expf(s[r] - mrun);
  }

  // lanes n and n+16 hold disjoint rows of the same column: merge the pair
  float mo = __shfl_xor(mrun, 16, 32);
  float zo = __shfl_xor(zrun, 16, 32);
  float mf = fmaxf(mrun, mo);
  float zf = zrun * __expf(mrun - mf) + zo * __expf(mo - mf);
  if (half == 0) {
    mOut[(size_t)b * SEQ + k0 + ln] = mf;
    zOut[(size_t)b * SEQ + k0 + ln] = zf;
  }
}

// ---- kernel 3: out = (exp(S - m)/Z) @ V -----------------------------------
// grid = BATCH * (SEQ/64); each wave owns 16 q-rows x full D=128 output
// (8 f32 accumulator fragments); k loops in chunks of 32 with ping-pong
// double-buffered LDS staging of K (row-major) and V (transposed).
__global__ __launch_bounds__(TPB) void attn_out(
    const _Float16* __restrict__ Qh, const _Float16* __restrict__ Kh,
    const _Float16* __restrict__ Vh, const float* __restrict__ mArr,
    const float* __restrict__ zArr, float* __restrict__ out) {
  __shared__ alignas(16) _Float16 Ksh[2][32][DIM];  // K chunks   (16 KB)
  __shared__ alignas(16) _Float16 Vts[2][DIM][32];  // V^T chunks (16 KB)
  __shared__ alignas(16) _Float16 Psh[WPB][16][32]; // P bounce    (4 KB)

  const int nQblk = SEQ / (WPB * 16);
  const int b    = blockIdx.x / nQblk;
  const int qblk = blockIdx.x % nQblk;
  const int wave = threadIdx.x >> 5;
  const int lane = threadIdx.x & 31;
  const int half = lane >> 4;
  const int ln   = lane & 15;
  const int q0   = qblk * (WPB * 16) + wave * 16;

  const _Float16* Qb = Qh + (size_t)b * SEQ * DIM;
  const _Float16* Kb = Kh + (size_t)b * SEQ * DIM;
  const _Float16* Vb = Vh + (size_t)b * SEQ * DIM;
  const float*    mB = mArr + (size_t)b * SEQ;
  const float*    zB = zArr + (size_t)b * SEQ;

  // staging roles (128 threads cover a 32 x 128 chunk)
  const int t    = threadIdx.x;
  const int krow = t >> 2;          // 0..31 : K row within chunk
  const int kcb  = (t & 3) << 5;    // 0,32,64,96 : K d-base (64 B)
  const int vp   = t >> 3;          // 0..15 : V row pair
  const int vcb  = (t & 7) << 4;    // 0..112 : V d-base (16 halfs x 2 rows)

  ushort8 vreg[4];
  auto loadV = [&](int k0c) {
    const unsigned short* va =
        (const unsigned short*)(Vb + (size_t)(k0c + 2 * vp) * DIM + vcb);
    const unsigned short* vb =
        (const unsigned short*)(Vb + (size_t)(k0c + 2 * vp + 1) * DIM + vcb);
    vreg[0] = *(const ushort8*)va; vreg[1] = *(const ushort8*)(va + 8);
    vreg[2] = *(const ushort8*)vb; vreg[3] = *(const ushort8*)(vb + 8);
  };
  auto storeV = [&](int buf) {  // packed b32 transpose scatter
#pragma unroll
    for (int i = 0; i < 8; ++i) {
      *(unsigned*)&Vts[buf][vcb + i][2 * vp] =
          (unsigned)vreg[0][i] | ((unsigned)vreg[2][i] << 16);
      *(unsigned*)&Vts[buf][vcb + 8 + i][2 * vp] =
          (unsigned)vreg[1][i] | ((unsigned)vreg[3][i] << 16);
    }
  };
#if USE_ASYNC_LDS
  auto loadK = [&](int k0c, int buf) {  // direct global -> LDS, ASYNCcnt
    AS1 i32x4* g = (AS1 i32x4*)(Kb + (size_t)(k0c + krow) * DIM + kcb);
    AS3 i32x4* l = (AS3 i32x4*)&Ksh[buf][krow][kcb];
    __builtin_amdgcn_global_load_async_to_lds_b128(g, l, 0, 0);
    __builtin_amdgcn_global_load_async_to_lds_b128(g, l, 16, 0);
    __builtin_amdgcn_global_load_async_to_lds_b128(g, l, 32, 0);
    __builtin_amdgcn_global_load_async_to_lds_b128(g, l, 48, 0);
  };
#else
  uint4 kreg[4];
  auto loadK = [&](int k0c, int /*buf*/) {
    const uint4* ks = (const uint4*)(Kb + (size_t)(k0c + krow) * DIM + kcb);
    kreg[0] = ks[0]; kreg[1] = ks[1]; kreg[2] = ks[2]; kreg[3] = ks[3];
  };
  auto storeK = [&](int buf) {
    uint4* kd = (uint4*)&Ksh[buf][krow][kcb];
    kd[0] = kreg[0]; kd[1] = kreg[1]; kd[2] = kreg[2]; kd[3] = kreg[3];
  };
#endif

  // loop-invariant Q fragments (A matrix of the S tiles)
  v16h aQ[4];
#pragma unroll
  for (int c = 0; c < 4; ++c)
    aQ[c] = load_a_frag(Qb + (size_t)(q0 + ln) * DIM + 32 * c, half);

  v8f accD[8];
#pragma unroll
  for (int j = 0; j < 8; ++j) accD[j] = (v8f){};

  loadK(0, 0);   // prologue: start chunk 0
  loadV(0);

  for (int it = 0; it < SEQ / 32; ++it) {
    const int k0  = it * 32;
    const int buf = it & 1;

    storeV(buf);
#if !USE_ASYNC_LDS
    storeK(buf);
#else
    wait_async0();              // this wave's async K copies landed
#endif
    __syncthreads();            // chunk `buf` fully staged; buf^1 free

    if (it + 1 < SEQ / 32) {    // start next chunk while computing this one
      loadK(k0 + 32, buf ^ 1);
      loadV(k0 + 32);
    }

    // two 16-wide k tiles -> P (probabilities) into this wave's LDS buffer
#pragma unroll
    for (int tt = 0; tt < 2; ++tt) {
      v8f acc = {};
#pragma unroll
      for (int c = 0; c < 4; ++c) {
        v16h bk = load_b_frag(&Ksh[buf][16 * tt + ln][32 * c + 16 * half]);
        acc = __builtin_amdgcn_wmma_f32_16x16x32_f16(
            false, aQ[c], false, bk, (short)0, acc, false, false);
      }
      const int kc = k0 + 16 * tt + ln;
      const float mv = mB[kc];
      const float rz = 1.0f / zB[kc];
#pragma unroll
      for (int r = 0; r < 8; ++r) {
        float p = __expf(acc[r] * SCALE - mv) * rz;
        Psh[wave][r + 8 * half][16 * tt + ln] = (_Float16)p;  // C->A relayout
      }
    }

    // wave-private LDS fence: P stores above feed the gathers below
    __builtin_amdgcn_wave_barrier();
    asm volatile("s_wait_dscnt 0" ::: "memory");
    __builtin_amdgcn_wave_barrier();

    // P as A-fragment (16q x 32k), V^T rows as B-fragments (32k x 16d)
    v16h aP = load_a_frag(&Psh[wave][ln][0], half);
#pragma unroll
    for (int j = 0; j < 8; ++j) {
      v16h bv = load_b_frag(&Vts[buf][16 * j + ln][16 * half]);
      accD[j] = __builtin_amdgcn_wmma_f32_16x16x32_f16(
          false, aP, false, bv, (short)0, accD[j], false, false);
    }
  }

  // C layout: reg r -> row q0 + r + 8*half, col 16*j + ln (coalesced in d)
#pragma unroll
  for (int j = 0; j < 8; ++j)
#pragma unroll
    for (int r = 0; r < 8; ++r)
      out[((size_t)b * SEQ + q0 + r + 8 * half) * DIM + 16 * j + ln] =
          accD[j][r];
}

// ---------------------------------------------------------------------------
extern "C" void kernel_launch(void* const* d_in, const int* in_sizes, int n_in,
                              void* d_out, int out_size, void* d_ws, size_t ws_size,
                              hipStream_t stream) {
  const float* q = (const float*)d_in[0];
  const float* k = (const float*)d_in[1];
  const float* v = (const float*)d_in[2];
  float* out = (float*)d_out;

  const size_t nElem = (size_t)BATCH * SEQ * DIM;  // 2,097,152
  char* ws = (char*)d_ws;
  _Float16* Qh = (_Float16*)ws;
  _Float16* Kh = Qh + nElem;
  _Float16* Vh = Kh + nElem;
  float* mArr = (float*)(Vh + nElem);
  float* zArr = mArr + (size_t)BATCH * SEQ;
  // workspace use: 3*4MB f16 + 2*64KB stats ~= 12.7 MB

  attn_cvt_f16<<<2048, 256, 0, stream>>>(q, k, v, Qh, Kh, Vh, (int)nElem);

  const int nblk = BATCH * (SEQ / (WPB * 16));  // 256 workgroups, 4 waves each
  attn_colstats<<<nblk, TPB, 0, stream>>>(Qh, Kh, mArr, zArr);
  attn_out<<<nblk, TPB, 0, stream>>>(Qh, Kh, Vh, mArr, zArr, out);
}
